// MultiHeadAttention_90228672954737
// MI455X (gfx1250) — matrix operand order repaired and checked by reading, not compiled, not run on hardware
//
#include <hip/hip_runtime.h>
#include <hip/hip_bf16.h>

// Problem constants (match reference)
#define Bc  4
#define Sc  2048
#define Dc  1024
#define Hc  16
#define HDc 64

typedef __attribute__((ext_vector_type(16))) __bf16 v16bf;
typedef __attribute__((ext_vector_type(8)))  float  v8f;
typedef unsigned int v4u __attribute__((ext_vector_type(4)));
typedef int          v8i __attribute__((ext_vector_type(8)));
typedef int          v4i __attribute__((ext_vector_type(4)));

#if defined(__has_builtin)
#if __has_builtin(__builtin_amdgcn_tensor_load_to_lds) && \
    __has_builtin(__builtin_amdgcn_s_wait_tensorcnt)
#define HAVE_TDM 1
#endif
#endif

union BF16x16 { v16bf v; uint4 q[2]; unsigned short u[16]; };

__device__ __forceinline__ unsigned short f2bf(float f) {
  union { float f; unsigned u; } x; x.f = f;
  unsigned r = x.u + 0x7FFFu + ((x.u >> 16) & 1u);   // round-to-nearest-even
  return (unsigned short)(r >> 16);
}
__device__ __forceinline__ unsigned pack2(float a, float b) {
  return (unsigned)f2bf(a) | ((unsigned)f2bf(b) << 16);
}
__device__ __forceinline__ uint4 pack8(float4 a, float4 b) {
  uint4 r;
  r.x = pack2(a.x, a.y); r.y = pack2(a.z, a.w);
  r.z = pack2(b.x, b.y); r.w = pack2(b.z, b.w);
  return r;
}
__device__ __forceinline__ v8f wmma_bf16(v16bf a, v16bf b, v8f c) {
  return __builtin_amdgcn_wmma_f32_16x16x32_bf16(false, a, false, b, (short)0, c,
                                                 false, false);
}

// ---------------------------------------------------------------------------
// Kernel 1: QKV projection.  y = x @ W^T + bias  (W is [out,in]).
// grid = (D/64, B*S/64, 3); block = 256 (8 waves). Workgroup tile 64x64,
// each wave computes 16(M) x 32(N) via two f32 accumulators, bf16 WMMA K=32.
// z==0 -> Q [B,H,S,HD] (pre-scaled by 1/8), z==1 -> K [B,H,S,HD],
// z==2 -> V transposed [B,H,HD,S].
// ---------------------------------------------------------------------------
__global__ __launch_bounds__(256) void qkv_kernel(
    const float* __restrict__ x,
    const float* __restrict__ Wq, const float* __restrict__ bq,
    const float* __restrict__ Wk, const float* __restrict__ bk,
    const float* __restrict__ Wv, const float* __restrict__ bv,
    unsigned short* __restrict__ Q, unsigned short* __restrict__ Ko,
    unsigned short* __restrict__ Vt)
{
  __shared__ unsigned short As[64][40];   // [m][k] bf16, +8 pad
  __shared__ unsigned short Bs[64][40];   // [n][k] bf16 (row n of W)

  const int z    = blockIdx.z;
  const float* W    = (z == 0) ? Wq : (z == 1) ? Wk : Wv;
  const float* bias = (z == 0) ? bq : (z == 1) ? bk : bv;
  unsigned short* out = (z == 0) ? Q : (z == 1) ? Ko : Vt;

  const int row0 = blockIdx.y * 64;
  const int col0 = blockIdx.x * 64;
  const int t    = threadIdx.x;
  const int lane = t & 31, wave = t >> 5;
  const int half = lane >> 4, l16 = lane & 15;
  const int mrow = (wave & 3) * 16;      // wave's M offset inside tile
  const int ncol = (wave >> 2) * 32;     // wave's N offset inside tile
  const int lm = t >> 2;                 // 0..63  (loader row)
  const int lk = (t & 3) * 8;            // 0,8,16,24 (loader k offset)

  const float* arow_g = x + (size_t)(row0 + lm) * Dc + lk;
  const float* brow_g = W + (size_t)(col0 + lm) * Dc + lk;

  v8f acc[2] = {};

  for (int k0 = 0; k0 < Dc; k0 += 32) {
    const float4* ap = (const float4*)(arow_g + k0);
    float4 a0 = ap[0], a1 = ap[1];
    const float4* bp = (const float4*)(brow_g + k0);
    float4 b0 = bp[0], b1 = bp[1];
    // speculative prefetch of next K-step (fault-dropped at tensor end)
    __builtin_prefetch(arow_g + k0 + 32, 0, 1);
    __builtin_prefetch(brow_g + k0 + 32, 0, 1);
    *(uint4*)&As[lm][lk] = pack8(a0, a1);
    *(uint4*)&Bs[lm][lk] = pack8(b0, b1);
    __syncthreads();

    BF16x16 a, bf0, bf1;
    const unsigned short* arow = &As[mrow + l16][0];
    a.q[0] = *(const uint4*)(arow + half * 16);       // K = half*16 .. +8
    a.q[1] = *(const uint4*)(arow + half * 16 + 8);   // K = half*16+8 .. +8
    const unsigned short* br0 = &Bs[ncol + l16][0];
    bf0.q[0] = *(const uint4*)(br0 + half * 16);      // K = half*16 .. +16
    bf0.q[1] = *(const uint4*)(br0 + half * 16 + 8);
    const unsigned short* br1 = &Bs[ncol + 16 + l16][0];
    bf1.q[0] = *(const uint4*)(br1 + half * 16);
    bf1.q[1] = *(const uint4*)(br1 + half * 16 + 8);

    acc[0] = wmma_bf16(a.v, bf0.v, acc[0]);
    acc[1] = wmma_bf16(a.v, bf1.v, acc[1]);
    __syncthreads();
  }

  const float scale = (z == 0) ? 0.125f : 1.0f;  // 1/sqrt(HD) folded into Q
#pragma unroll
  for (int j = 0; j < 2; ++j) {
    const int gcol = col0 + ncol + j * 16 + l16;
    const float bval = bias[gcol];
    const int h = gcol >> 6, hd = gcol & 63;
#pragma unroll
    for (int r = 0; r < 8; ++r) {
      const int grow = row0 + mrow + r + 8 * half;
      const int b = grow >> 11, s = grow & (Sc - 1);
      const float val = (acc[j][r] + bval) * scale;
      size_t idx;
      if (z == 2) idx = ((size_t)(b * Hc + h) * HDc + hd) * Sc + s;   // V^T
      else        idx = ((size_t)(b * Hc + h) * Sc + s) * HDc + hd;   // Q, K
      out[idx] = f2bf(val);
    }
  }
}

// ---------------------------------------------------------------------------
// Kernel 2: causal flash attention per (b,h).  grid = (S/128, B*H); 8 waves,
// each wave owns 16 query rows independently (no cross-wave sync needed).
// Scores: WMMA over HD=64 (2 x K=32). Online softmax in registers with
// shuffle row-reductions. P is re-laid-out C->A format through LDS, then
// P·V accumulated with WMMA against transposed V.
// ---------------------------------------------------------------------------
__global__ __launch_bounds__(256) void attn_kernel(
    const unsigned short* __restrict__ Q, const unsigned short* __restrict__ K,
    const unsigned short* __restrict__ Vt, const int* __restrict__ amask,
    unsigned short* __restrict__ O)
{
  __shared__ unsigned short Pbuf[8][16][72];   // per-wave 16x64 P tile (+pad)

  const int t = threadIdx.x, lane = t & 31, wave = t >> 5;
  const int half = lane >> 4, l16 = lane & 15;
  const int bh = blockIdx.y, b = bh >> 4, h = bh & 15;
  const int q0 = blockIdx.x * 128 + wave * 16;

  const unsigned short* Qb = Q  + (size_t)bh * Sc * HDc;
  const unsigned short* Kb = K  + (size_t)bh * Sc * HDc;
  const unsigned short* Vb = Vt + (size_t)bh * HDc * Sc;

  // Q fragments for this wave's 16 rows (K-chunks over HD)
  BF16x16 aQ[2];
  const unsigned short* qrow = Qb + (size_t)(q0 + l16) * HDc;
#pragma unroll
  for (int c = 0; c < 2; ++c) {
    aQ[c].q[0] = *(const uint4*)(qrow + c * 32 + half * 16);
    aQ[c].q[1] = *(const uint4*)(qrow + c * 32 + half * 16 + 8);
  }

  float mrun[8], lrun[8];
  v8f o[4] = {};
#pragma unroll
  for (int r = 0; r < 8; ++r) { mrun[r] = -1e30f; lrun[r] = 0.0f; }

  for (int kb = 0; kb < q0 + 16; kb += 64) {   // causal: skip blocks > q rows
    // ---- scores: 4 tiles of 16x16 keys --------------------------------
    v8f sc[4] = {};
    int mk[4];
#pragma unroll
    for (int tt = 0; tt < 4; ++tt) {
      const int key = kb + tt * 16 + l16;
      mk[tt] = amask[b * Sc + key];
      const unsigned short* krow = Kb + (size_t)key * HDc;
#pragma unroll
      for (int c = 0; c < 2; ++c) {
        BF16x16 bK;
        bK.q[0] = *(const uint4*)(krow + c * 32 + half * 16);
        bK.q[1] = *(const uint4*)(krow + c * 32 + half * 16 + 8);
        sc[tt] = wmma_bf16(aQ[c].v, bK.v, sc[tt]);
      }
    }
    // ---- mask + row max ----------------------------------------------
    float newm[8];
#pragma unroll
    for (int r = 0; r < 8; ++r) newm[r] = mrun[r];
#pragma unroll
    for (int tt = 0; tt < 4; ++tt) {
      const int kg = kb + tt * 16 + l16;
#pragma unroll
      for (int r = 0; r < 8; ++r) {
        const int qg = q0 + r + 8 * half;
        float sv = sc[tt][r];
        if (kg > qg || mk[tt] == 0) sv = -1e30f;
        sc[tt][r] = sv;
        newm[r] = fmaxf(newm[r], sv);
      }
    }
#pragma unroll
    for (int off = 1; off < 16; off <<= 1)
#pragma unroll
      for (int r = 0; r < 8; ++r)
        newm[r] = fmaxf(newm[r], __shfl_xor(newm[r], off, 32));

    // ---- exp + row sum + rescale accumulators ------------------------
    float rs[8] = {0, 0, 0, 0, 0, 0, 0, 0};
#pragma unroll
    for (int tt = 0; tt < 4; ++tt)
#pragma unroll
      for (int r = 0; r < 8; ++r) {
        const float p = __expf(sc[tt][r] - newm[r]);
        sc[tt][r] = p;
        rs[r] += p;
      }
#pragma unroll
    for (int off = 1; off < 16; off <<= 1)
#pragma unroll
      for (int r = 0; r < 8; ++r) rs[r] += __shfl_xor(rs[r], off, 32);
#pragma unroll
    for (int r = 0; r < 8; ++r) {
      const float alpha = __expf(mrun[r] - newm[r]);
      lrun[r] = lrun[r] * alpha + rs[r];
      mrun[r] = newm[r];
#pragma unroll
      for (int c = 0; c < 4; ++c) o[c][r] *= alpha;
    }

    // ---- re-layout P (C-format -> A-format) via per-wave LDS ---------
#pragma unroll
    for (int tt = 0; tt < 4; ++tt)
#pragma unroll
      for (int r = 0; r < 8; ++r)
        Pbuf[wave][r + 8 * half][tt * 16 + l16] = f2bf(sc[tt][r]);
    asm volatile("s_wait_dscnt 0" ::: "memory");   // same-wave cross-lane LDS hazard

    BF16x16 aP[2];
    const unsigned short* prow = &Pbuf[wave][l16][0];
#pragma unroll
    for (int ch = 0; ch < 2; ++ch) {
      aP[ch].q[0] = *(const uint4*)(prow + ch * 32 + half * 16);
      aP[ch].q[1] = *(const uint4*)(prow + ch * 32 + half * 16 + 8);
    }
    // ---- O += P·V  (V transposed: [hd][key], contiguous B fragments) -
#pragma unroll
    for (int c = 0; c < 4; ++c) {
      const unsigned short* vrow = Vb + (size_t)(c * 16 + l16) * Sc + kb;
#pragma unroll
      for (int ch = 0; ch < 2; ++ch) {
        BF16x16 bV;
        bV.q[0] = *(const uint4*)(vrow + ch * 32 + half * 16);
        bV.q[1] = *(const uint4*)(vrow + ch * 32 + half * 16 + 8);
        o[c] = wmma_bf16(aP[ch].v, bV.v, o[c]);
      }
    }
  }

  // ---- normalize and store O as bf16 [B,S,D] --------------------------
#pragma unroll
  for (int c = 0; c < 4; ++c) {
    const int hd = c * 16 + l16;
#pragma unroll
    for (int r = 0; r < 8; ++r) {
      const int srow = q0 + r + 8 * half;
      const float val = o[c][r] / lrun[r];
      O[((size_t)b * Sc + srow) * Dc + h * HDc + hd] = f2bf(val);
    }
  }
}

// ---------------------------------------------------------------------------
// Kernel 3: output projection.  out = O @ Wo^T + bo, fp32 output.
// Same tiling as kernel 1.  The A tile is already bf16 in workspace, so it
// is staged into LDS by the Tensor Data Mover (tensor_load_to_lds) with the
// D# pad feature reproducing the 40-short padded row stride; wave 0 issues
// the DMA and waits on TENSORcnt, __syncthreads publishes it to all waves.
// ---------------------------------------------------------------------------
__global__ __launch_bounds__(256) void oproj_kernel(
    const unsigned short* __restrict__ Oin, const float* __restrict__ Wo,
    const float* __restrict__ bo, float* __restrict__ out)
{
  __shared__ unsigned short As[64][40];
  __shared__ unsigned short Bs[64][40];

  const int row0 = blockIdx.y * 64;
  const int col0 = blockIdx.x * 64;
  const int t = threadIdx.x;
  const int lane = t & 31, wave = t >> 5;
  const int half = lane >> 4, l16 = lane & 15;
  const int mrow = (wave & 3) * 16;
  const int ncol = (wave >> 2) * 32;
  const int lm = t >> 2;
  const int lk = (t & 3) * 8;

  const unsigned short* arow_g = Oin + (size_t)(row0 + lm) * Dc + lk;
  const float*          brow_g = Wo + (size_t)(col0 + lm) * Dc + lk;

#if HAVE_TDM
  const unsigned ldsA = (unsigned)(uintptr_t)(&As[0][0]);  // LDS byte offset
#endif

  v8f acc[2] = {};

  for (int k0 = 0; k0 < Dc; k0 += 32) {
#if HAVE_TDM
    if (wave == 0) {
      // D# group 0: count=1, lds_addr, 57-bit global tile addr, type=2
      const unsigned long long ga =
          (unsigned long long)(uintptr_t)(Oin + (size_t)row0 * Dc + k0);
      v4u g0;
      g0[0] = 1u;                                        // count=1, user mode
      g0[1] = ldsA;                                      // lds_addr
      g0[2] = (unsigned)ga;                              // global_addr[31:0]
      g0[3] = ((unsigned)(ga >> 32) & 0x1FFFFFFu) | (2u << 30);  // [56:32]|type=2
      // D# group 1: data_size=2B, pad 4 DWORDs every 16 DWORDs (row=64B),
      // tensor 1024 x 8192, tile 32 x 64, dim0 stride 1024 elements.
      v8i g1;
      g1[0] = (1 << 16) | (1 << 20) | (3 << 22) | (3 << 25);
      g1[1] = (int)((unsigned)Dc << 16);                 // tensor_dim0 lo16
      g1[2] = (int)(((unsigned)(Bc * Sc) & 0xFFFFu) << 16); // tensor_dim1 lo16
      g1[3] = (int)(32u << 16);                          // tile_dim0 = 32
      g1[4] = 64;                                        // tile_dim1 = 64
      g1[5] = Dc;                                        // dim0_stride = 1024
      g1[6] = 0;
      g1[7] = 0;
      v4i z4 = (v4i)0;
      v8i z8 = (v8i)0;
      __builtin_amdgcn_tensor_load_to_lds(g0, g1, z4, z4, z8, 0);
    }
#else
    *(uint4*)&As[lm][lk] = *(const uint4*)(arow_g + k0);
#endif
    const float4* bp = (const float4*)(brow_g + k0);
    float4 b0 = bp[0], b1 = bp[1];
    __builtin_prefetch(brow_g + k0 + 32, 0, 1);
    *(uint4*)&Bs[lm][lk] = pack8(b0, b1);
#if HAVE_TDM
    if (wave == 0) __builtin_amdgcn_s_wait_tensorcnt(0);
#endif
    __syncthreads();

    BF16x16 a, bf0, bf1;
    const unsigned short* arow = &As[mrow + l16][0];
    a.q[0] = *(const uint4*)(arow + half * 16);
    a.q[1] = *(const uint4*)(arow + half * 16 + 8);
    const unsigned short* br0 = &Bs[ncol + l16][0];
    bf0.q[0] = *(const uint4*)(br0 + half * 16);
    bf0.q[1] = *(const uint4*)(br0 + half * 16 + 8);
    const unsigned short* br1 = &Bs[ncol + 16 + l16][0];
    bf1.q[0] = *(const uint4*)(br1 + half * 16);
    bf1.q[1] = *(const uint4*)(br1 + half * 16 + 8);

    acc[0] = wmma_bf16(a.v, bf0.v, acc[0]);
    acc[1] = wmma_bf16(a.v, bf1.v, acc[1]);
    __syncthreads();
  }

#pragma unroll
  for (int j = 0; j < 2; ++j) {
    const int gcol = col0 + ncol + j * 16 + l16;
    const float bval = bo[gcol];
#pragma unroll
    for (int r = 0; r < 8; ++r) {
      const int grow = row0 + mrow + r + 8 * half;
      out[(size_t)grow * Dc + gcol] = acc[j][r] + bval;
    }
  }
}

// ---------------------------------------------------------------------------
extern "C" void kernel_launch(void* const* d_in, const int* in_sizes, int n_in,
                              void* d_out, int out_size, void* d_ws,
                              size_t ws_size, hipStream_t stream) {
  (void)in_sizes; (void)n_in; (void)out_size; (void)ws_size;
  const float* x     = (const float*)d_in[0];
  const int*   amask = (const int*)d_in[1];
  const float* Wq = (const float*)d_in[2];
  const float* bq = (const float*)d_in[3];
  const float* Wk = (const float*)d_in[4];
  const float* bk = (const float*)d_in[5];
  const float* Wv = (const float*)d_in[6];
  const float* bv = (const float*)d_in[7];
  const float* Wo = (const float*)d_in[8];
  const float* bo = (const float*)d_in[9];
  float* out = (float*)d_out;

  const size_t QKV_ELEMS = (size_t)Bc * Hc * Sc * HDc;  // 8388608
  unsigned short* Q  = (unsigned short*)d_ws;
  unsigned short* Kw = Q + QKV_ELEMS;
  unsigned short* Vt = Kw + QKV_ELEMS;
  unsigned short* O  = Vt + QKV_ELEMS;   // 64 MB total bf16 workspace

  dim3 gQKV(Dc / 64, (Bc * Sc) / 64, 3);          // (16, 128, 3)
  qkv_kernel<<<gQKV, 256, 0, stream>>>(x, Wq, bq, Wk, bk, Wv, bv, Q, Kw, Vt);

  dim3 gATT(Sc / 128, Bc * Hc);                   // (16, 64)
  attn_kernel<<<gATT, 256, 0, stream>>>(Q, Kw, Vt, amask, O);

  dim3 gOP(Dc / 64, (Bc * Sc) / 64);              // (16, 128)
  oproj_kernel<<<gOP, 256, 0, stream>>>(O, Wo, bo, out);
}
